// InfoNCELoss_80427557585044
// MI455X (gfx1250) — matrix operand order repaired and checked by reading, not compiled, not run on hardware
//
#include <hip/hip_runtime.h>

// ---------- types for WMMA ----------
typedef __bf16 v16bf __attribute__((ext_vector_type(16)));
typedef __bf16 v8bf  __attribute__((ext_vector_type(8)));
typedef float  v8f   __attribute__((ext_vector_type(8)));

union BFrag { v16bf v; v8bf h[2]; };

#define D_DIM   512
#define KFRAGS  16          // 512 / 32
#define LDS_STRIDE 520      // 512 + 8 pad elements -> 1040B row stride, conflict-free
#define INV_T   (1.0f / 0.07f)

#if __has_builtin(__builtin_amdgcn_global_load_async_to_lds_b128) && \
    __has_builtin(__builtin_amdgcn_s_wait_asynccnt)
#define USE_ASYNC_LDS 1
// builtin expects: (int4 AS1*, int4 AS3*, imm offset, imm cpol)
typedef int v4i_vs __attribute__((__vector_size__(16)));
typedef __attribute__((address_space(1))) v4i_vs gv4i;
typedef __attribute__((address_space(3))) v4i_vs lv4i;
#endif

__device__ __forceinline__ unsigned short f2bf(float f) {
  unsigned int u = __float_as_uint(f);
  unsigned int rnd = 0x7FFFu + ((u >> 16) & 1u);   // round to nearest even
  return (unsigned short)((u + rnd) >> 16);
}

__device__ __forceinline__ void load_bfrag(BFrag& b, const unsigned short* p, int kf) {
  b.h[0] = *(const v8bf*)(p + kf * 32);
  b.h[1] = *(const v8bf*)(p + kf * 32 + 16);
}

// ---------------------------------------------------------------------------
// Kernel 1: L2-normalize rows of X (fp32, D=512) and store bf16 into Y.
// ---------------------------------------------------------------------------
__global__ __launch_bounds__(256) void nce_norm_bf16(const float* __restrict__ X,
                                                     unsigned short* __restrict__ Y,
                                                     int Bn) {
  const int wave = threadIdx.x >> 5;
  const int lane = threadIdx.x & 31;
  const int row  = blockIdx.x * 8 + wave;
  if (row >= Bn) return;

  const float* xr = X + (size_t)row * D_DIM;
  float4 v[4];
  float ss = 0.0f;
#pragma unroll
  for (int c = 0; c < 4; ++c) {
    v[c] = *(const float4*)(xr + c * 128 + lane * 4);
    ss += v[c].x * v[c].x + v[c].y * v[c].y + v[c].z * v[c].z + v[c].w * v[c].w;
  }
#pragma unroll
  for (int m = 16; m >= 1; m >>= 1) ss += __shfl_xor(ss, m, 32);

  const float scale = 1.0f / fmaxf(sqrtf(ss), 1e-12f);

  unsigned short* yr = Y + (size_t)row * D_DIM;
#pragma unroll
  for (int c = 0; c < 4; ++c) {
    ushort4 o;
    o.x = f2bf(v[c].x * scale);
    o.y = f2bf(v[c].y * scale);
    o.z = f2bf(v[c].z * scale);
    o.w = f2bf(v[c].w * scale);
    *(ushort4*)(yr + c * 128 + lane * 4) = o;
  }
}

// ---------------------------------------------------------------------------
// Kernel 2: fused sim = f1n @ f2n^T / T, exp, masked row sums.
// 8 waves/block; each wave owns 16 rows (A resident, K=512) and computes TWO
// 16-col subtiles at once (independent WMMA chains) with double-buffered B.
// f2 streamed through LDS in 32-row tiles (async global->LDS if available).
// ---------------------------------------------------------------------------
__global__ __launch_bounds__(256) void nce_wmma_sums(const unsigned short* __restrict__ F1,
                                                     const unsigned short* __restrict__ F2,
                                                     const int* __restrict__ labels,
                                                     float* __restrict__ pos_sum,
                                                     float* __restrict__ tot_sum,
                                                     int Bn) {
  __shared__ unsigned short tileB[32 * LDS_STRIDE];

  const int t     = threadIdx.x;
  const int wave  = t >> 5;
  const int lane  = t & 31;
  const int lrow  = lane & 15;   // N (for B/C) or M (for A) sub-index
  const int lhalf = lane >> 4;   // which K half-range this lane carries
  const int m0    = blockIdx.x * 128 + wave * 16;

  // --- A fragments: rows m0..m0+15, all of K=512, resident in registers ---
  // ISA 16-bit A 16x32 layout: lane L (M = L%16) holds K {kb..kb+7} U {kb+16..kb+23},
  // kb = kf*32 + (L/16)*8  -> two contiguous b128 loads per fragment.
  BFrag a[KFRAGS];
  {
    const unsigned short* base = F1 + (size_t)(m0 + lrow) * D_DIM;
#pragma unroll
    for (int kf = 0; kf < KFRAGS; ++kf) {
      const int kb = kf * 32 + lhalf * 8;
      a[kf].h[0] = *(const v8bf*)(base + kb);
      a[kf].h[1] = *(const v8bf*)(base + kb + 16);
    }
  }

  int labi[8];
#pragma unroll
  for (int r = 0; r < 8; ++r) labi[r] = labels[m0 + r + 8 * lhalf];

  float tot[8], pos[8];
#pragma unroll
  for (int r = 0; r < 8; ++r) { tot[r] = 0.0f; pos[r] = 0.0f; }

  // staging addresses for this thread: 128B (64 elems) per thread per tile
  const int srow = t >> 3;         // 0..31
  const int scb  = (t & 7) * 64;   // col base

  for (int j0 = 0; j0 < Bn; j0 += 32) {
    __syncthreads();
    // ---- stage f2n[j0 .. j0+31][0..511] into LDS ----
    {
      const unsigned short* src = F2 + (size_t)(j0 + srow) * D_DIM + scb;
      unsigned short* dst = tileB + srow * LDS_STRIDE + scb;
#ifdef USE_ASYNC_LDS
#pragma unroll
      for (int q = 0; q < 8; ++q)
        __builtin_amdgcn_global_load_async_to_lds_b128(
            (gv4i*)(src + q * 8), (lv4i*)(dst + q * 8), 0, 0);
#else
#pragma unroll
      for (int q = 0; q < 8; ++q)
        *(uint4*)(dst + q * 8) = *(const uint4*)(src + q * 8);
      if (j0 + 32 < Bn)
        __builtin_prefetch(src + 32 * D_DIM, 0, 0);   // global_prefetch_b8
#endif
    }
#ifdef USE_ASYNC_LDS
    __builtin_amdgcn_s_wait_asynccnt(0);
#endif
    __syncthreads();

    // ISA 16-bit B 32x16 layout: lane L (N = L%16) holds K {kb..kb+7} U {kb+16..kb+23},
    // kb = kf*32 + (L/16)*8  -> two b128 LDS loads.
    const unsigned short* brow0 = tileB + lrow * LDS_STRIDE + lhalf * 8;
    const unsigned short* brow1 = brow0 + 16 * LDS_STRIDE;

    v8f c0 = {}, c1 = {};
    BFrag b0[2], b1[2];
    load_bfrag(b0[0], brow0, 0);
    load_bfrag(b1[0], brow1, 0);
#pragma unroll
    for (int kf = 0; kf < KFRAGS; ++kf) {
      const int cur = kf & 1, nxt = cur ^ 1;
      if (kf + 1 < KFRAGS) {               // prefetch next B frags into other buffer
        load_bfrag(b0[nxt], brow0, kf + 1);
        load_bfrag(b1[nxt], brow1, kf + 1);
      }
      c0 = __builtin_amdgcn_wmma_f32_16x16x32_bf16(
               false, a[kf].v, false, b0[cur].v, (short)0, c0, false, false);
      c1 = __builtin_amdgcn_wmma_f32_16x16x32_bf16(
               false, a[kf].v, false, b1[cur].v, (short)0, c1, false, false);
    }

    const int labj0 = labels[j0 + lrow];
    const int labj1 = labels[j0 + 16 + lrow];
#pragma unroll
    for (int r = 0; r < 8; ++r) {
      const float e0 = __expf(c0[r] * INV_T);     // v_exp_f32
      const float e1 = __expf(c1[r] * INV_T);
      tot[r] += e0 + e1;
      pos[r] += ((labi[r] == labj0) ? e0 : 0.0f) + ((labi[r] == labj1) ? e1 : 0.0f);
    }
  }

  // ---- reduce over N (16 lanes within each half; xor masks stay in-half) ----
#pragma unroll
  for (int r = 0; r < 8; ++r) {
#pragma unroll
    for (int m = 1; m <= 8; m <<= 1) {
      tot[r] += __shfl_xor(tot[r], m, 32);
      pos[r] += __shfl_xor(pos[r], m, 32);
    }
  }
  if (lrow == 0) {
#pragma unroll
    for (int r = 0; r < 8; ++r) {
      const int row = m0 + r + 8 * lhalf;   // wave exclusively owns its 16 rows
      tot_sum[row] = tot[r];
      pos_sum[row] = pos[r];
    }
  }
}

// ---------------------------------------------------------------------------
// Kernel 3: loss = mean( log(tot) - log(pos) )
// ---------------------------------------------------------------------------
__global__ __launch_bounds__(256) void nce_final_loss(const float* __restrict__ pos_sum,
                                                      const float* __restrict__ tot_sum,
                                                      float* __restrict__ out, int Bn) {
  __shared__ float red[256];
  float acc = 0.0f;
  for (int i = threadIdx.x; i < Bn; i += 256)
    acc += __logf(tot_sum[i]) - __logf(pos_sum[i]);
  red[threadIdx.x] = acc;
  __syncthreads();
  for (int s = 128; s > 0; s >>= 1) {
    if (threadIdx.x < s) red[threadIdx.x] += red[threadIdx.x + s];
    __syncthreads();
  }
  if (threadIdx.x == 0) out[0] = red[0] / (float)Bn;
}

// ---------------------------------------------------------------------------
extern "C" void kernel_launch(void* const* d_in, const int* in_sizes, int n_in,
                              void* d_out, int out_size, void* d_ws, size_t ws_size,
                              hipStream_t stream) {
  const float* f1     = (const float*)d_in[0];
  const float* f2     = (const float*)d_in[1];
  const int*   labels = (const int*)d_in[2];

  const int Bn = in_sizes[2];                 // 8192
  // workspace layout: [f1n bf16 B*D][f2n bf16 B*D][pos f32 B][tot f32 B]
  unsigned short* f1n = (unsigned short*)d_ws;
  unsigned short* f2n = f1n + (size_t)Bn * D_DIM;
  float* pos = (float*)(f2n + (size_t)Bn * D_DIM);
  float* tot = pos + Bn;

  nce_norm_bf16<<<Bn / 8, 256, 0, stream>>>(f1, f1n, Bn);
  nce_norm_bf16<<<Bn / 8, 256, 0, stream>>>(f2, f2n, Bn);
  nce_wmma_sums<<<Bn / 128, 256, 0, stream>>>(f1n, f2n, labels, pos, tot, Bn);
  nce_final_loss<<<1, 256, 0, stream>>>(pos, tot, (float*)d_out, Bn);
}
